// ContextEncoder_89352499626119
// MI455X (gfx1250) — matrix operand order repaired and checked by reading, not compiled, not run on hardware
//
#include <hip/hip_runtime.h>
#include <hip/hip_bf16.h>

// ---------------- constants from the reference ----------------
#define NG        64
#define HID       32
#define FIL       64
#define NLAYERS   5
#define CUTOFF_F  10.0f
#define PI_F      3.14159265358979f
#define LOG2_F    0.69314718055994531f

typedef __attribute__((ext_vector_type(16))) _Float16 v16h;
typedef __attribute__((ext_vector_type(8)))  _Float16 v8h;
typedef __attribute__((ext_vector_type(8)))  float    v8f;

__device__ __forceinline__ float ssp(float x) {
    // shifted softplus: log(1 + exp(x)) - log(2)
    return __logf(1.0f + __expf(x)) - LOG2_F;
}

// ---------------- init: h = node_attr ----------------
__global__ void init_h_kernel(const float* __restrict__ na, float* __restrict__ h, int n) {
    int i = blockIdx.x * blockDim.x + threadIdx.x;
    if (i < n) h[i] = na[i];
}

// ---------------- per-edge geometry: edge_length + cosine gate ----------------
__global__ void edge_geom_kernel(const float* __restrict__ pos, const int* __restrict__ ei,
                                 float* __restrict__ EL, float* __restrict__ CG, int E) {
    int e = blockIdx.x * blockDim.x + threadIdx.x;
    if (e >= E) return;
    int s = ei[e];
    int d = ei[E + e];
    float dx = pos[3 * s + 0] - pos[3 * d + 0];
    float dy = pos[3 * s + 1] - pos[3 * d + 1];
    float dz = pos[3 * s + 2] - pos[3 * d + 2];
    float len = __fsqrt_rn(dx * dx + dy * dy + dz * dz);
    float c = 0.5f * (__cosf(len * (PI_F / CUTOFF_F)) + 1.0f);
    c = (len <= CUTOFF_F) ? c : 0.0f;
    EL[e] = len;
    CG[e] = c;
}

// ---------------- xf = h @ lin1_w[l]  (and zero agg) ----------------
__global__ void xf_zero_kernel(const float* __restrict__ h, const float* __restrict__ w1,
                               float* __restrict__ xf, float* __restrict__ agg, int n) {
    int idx = blockIdx.x * blockDim.x + threadIdx.x;
    if (idx >= n * FIL) return;
    int node = idx >> 6;
    int f    = idx & 63;
    const float* hr = &h[node * HID];
    const float* wc = &w1[f];
    float acc = 0.0f;
#pragma unroll
    for (int k = 0; k < HID; ++k) acc += hr[k] * wc[k * FIL];
    xf[idx]  = acc;
    agg[idx] = 0.0f;
}

// ---------------- fused edge kernel: Gaussian -> MLP (WMMA) -> gate -> gather -> scatter-add ----
// persistent: one wave32 per 16-edge tile per iteration; 8 waves / block; uniform trip count
__global__ void __launch_bounds__(256)
edge_mlp_kernel(const float* __restrict__ EL, const float* __restrict__ CG,
                const int* __restrict__ ei,
                const float* __restrict__ dn1w, const float* __restrict__ dn1b,
                const float* __restrict__ dn2w, const float* __restrict__ dn2b,
                const float* __restrict__ xf, float* __restrict__ agg,
                int E, int ntiles, int iters, int totalWaves) {
    // weights transposed to [N][K] f16 so B-fragments are contiguous per lane
    __shared__ __align__(16) _Float16 w1T[FIL * NG];
    __shared__ __align__(16) _Float16 w2T[FIL * FIL];
    __shared__ float b1s[FIL], b2s[FIL];
    // per-wave transpose scratch: 16 rows x (64+8 pad) halves
    __shared__ __align__(16) _Float16 scratch[8 * 16 * 72];

    const int t = threadIdx.x;
    for (int i = t; i < FIL * NG; i += 256) {
        int k = i >> 6, n = i & 63;
        w1T[n * NG + k]  = (_Float16)dn1w[i];   // dn1w is [K=NG][N=FIL] row-major
        w2T[n * FIL + k] = (_Float16)dn2w[i];
    }
    for (int i = t; i < FIL; i += 256) { b1s[i] = dn1b[i]; b2s[i] = dn2b[i]; }
    __syncthreads();

    const int wave = t >> 5;
    const int lane = t & 31;
    const int waveGlobal = blockIdx.x * 8 + wave;

    const int mrow = lane & 15;   // matrix row owned by this lane (A) / column (B,D)
    const int hi   = lane >> 4;   // lane half

    const float delta = CUTOFF_F / (float)(NG - 1);
    const float coeff = -0.5f / (delta * delta);

    _Float16* sc = &scratch[wave * 16 * 72];

    for (int it = 0; it < iters; ++it) {           // uniform trip count for all waves
        const int tile  = waveGlobal + it * totalWaves;
        const bool valid = (tile < ntiles);
        const int tclamp = valid ? tile : 0;
        const int ebase  = tclamp * 16;

        // edge length for this lane's A-row
        const int eA = min(ebase + mrow, E - 1);
        const float len = EL[eA];

        // ---- A fragments (Gaussian smearing), built directly in contiguous k order (k = 16*hi + i), same as B ----
        v16h aF[2];
#pragma unroll
        for (int ks = 0; ks < 2; ++ks) {
            const int k0 = ks * 32 + hi * 16;
#pragma unroll
            for (int i = 0; i < 16; ++i) {
                const int kk = k0 + i;
                const float d = len - (float)kk * delta;
                aF[ks][i] = (_Float16)__expf(coeff * d * d);
            }
        }

        // ---- GEMM1: edge_attr[16x64] @ dn1[64x64] ----
        v8f acc1[4];
#pragma unroll
        for (int nt = 0; nt < 4; ++nt) {
            v8f c = {0.f, 0.f, 0.f, 0.f, 0.f, 0.f, 0.f, 0.f};
            const int ncol = nt * 16 + mrow;
#pragma unroll
            for (int ks = 0; ks < 2; ++ks) {
                const int kb = ks * 32 + hi * 16;
                v16h b = *(const v16h*)&w1T[ncol * NG + kb];
                c = __builtin_amdgcn_wmma_f32_16x16x32_f16(false, aF[ks], false, b,
                                                           (short)0, c, false, false);
            }
            acc1[nt] = c;
        }

        // ---- bias + shifted softplus, bounce through LDS to re-layout D -> A ----
#pragma unroll
        for (int nt = 0; nt < 4; ++nt) {
            const int ncol = nt * 16 + mrow;
            const float bias = b1s[ncol];
#pragma unroll
            for (int r = 0; r < 8; ++r) {
                const int M = r + hi * 8;
                sc[M * 72 + ncol] = (_Float16)ssp(acc1[nt][r] + bias);
            }
        }
        __syncthreads();  // cross-lane LDS dependency within each wave

        v16h a2[2];
#pragma unroll
        for (int ks = 0; ks < 2; ++ks) {
            const int k0 = ks * 32 + hi * 16;
            v8h lo  = *(const v8h*)&sc[mrow * 72 + k0];
            v8h hi8 = *(const v8h*)&sc[mrow * 72 + k0 + 8];
#pragma unroll
            for (int i = 0; i < 8; ++i) { a2[ks][i] = lo[i]; a2[ks][i + 8] = hi8[i]; }
        }

        // ---- GEMM2: ssp(...)[16x64] @ dn2[64x64] ----
        v8f acc2[4];
#pragma unroll
        for (int nt = 0; nt < 4; ++nt) {
            v8f c = {0.f, 0.f, 0.f, 0.f, 0.f, 0.f, 0.f, 0.f};
            const int ncol = nt * 16 + mrow;
#pragma unroll
            for (int ks = 0; ks < 2; ++ks) {
                const int kb = ks * 32 + hi * 16;
                v16h b = *(const v16h*)&w2T[ncol * FIL + kb];
                c = __builtin_amdgcn_wmma_f32_16x16x32_f16(false, a2[ks], false, b,
                                                           (short)0, c, false, false);
            }
            acc2[nt] = c;
        }

        // ---- epilogue: bias, cosine gate, gather xf[src], scatter-add agg[dst] ----
#pragma unroll
        for (int r = 0; r < 8; ++r) {
            const int M = r + hi * 8;
            const int e = ebase + M;
            const bool ok = valid && (e < E);
            const int ec = ok ? e : 0;
            const float cg = CG[ec];
            const int s = ei[ec];
            const int d = ei[E + ec];
#pragma unroll
            for (int nt = 0; nt < 4; ++nt) {
                const int n = nt * 16 + mrow;
                float v = (acc2[nt][r] + b2s[n]) * cg;
                v *= xf[s * FIL + n];
                if (ok) atomicAdd(&agg[d * FIL + n], v);
            }
        }
        __syncthreads();  // protect wave-private scratch across iterations (uniform)
    }
}

// ---------------- node update: conv = agg@lin2 + b; h += ssp(conv)@lin + b ----------------
// one wave32 per node; lane = output channel
__global__ void __launch_bounds__(256)
node_update_kernel(const float* __restrict__ agg,
                   const float* __restrict__ l2w, const float* __restrict__ l2b,
                   const float* __restrict__ lw,  const float* __restrict__ lb,
                   float* __restrict__ h, int n) {
    const int node = blockIdx.x * 8 + (threadIdx.x >> 5);
    const int c = threadIdx.x & 31;
    if (node >= n) return;
    const float* ar = &agg[node * FIL];
    float acc = l2b[c];
#pragma unroll
    for (int k = 0; k < FIL; ++k) acc += ar[k] * l2w[k * HID + c];
    const float s = ssp(acc);
    float o = lb[c];
#pragma unroll
    for (int k = 0; k < HID; ++k) {
        const float sk = __shfl(s, k, 32);
        o += sk * lw[k * HID + c];
    }
    h[node * HID + c] += o;
}

// ---------------- launcher ----------------
extern "C" void kernel_launch(void* const* d_in, const int* in_sizes, int n_in,
                              void* d_out, int out_size, void* d_ws, size_t ws_size,
                              hipStream_t stream) {
    const float* node_attr = (const float*)d_in[0];
    const float* pos       = (const float*)d_in[1];
    const int*   ei        = (const int*)d_in[2];
    const float* dn1w      = (const float*)d_in[3];
    const float* dn1b      = (const float*)d_in[4];
    const float* dn2w      = (const float*)d_in[5];
    const float* dn2b      = (const float*)d_in[6];
    const float* lin1w     = (const float*)d_in[7];
    const float* lin2w     = (const float*)d_in[8];
    const float* lin2b     = (const float*)d_in[9];
    const float* linw      = (const float*)d_in[10];
    const float* linb      = (const float*)d_in[11];

    const int N = in_sizes[0] / HID;
    const int E = in_sizes[2] / 2;

    // workspace: EL[E] f32, CG[E] f32, XF[N*64] f32, AGG[N*64] f32  (~64.3 MB)
    char* p = (char*)d_ws;
    float* EL  = (float*)p; p += ((size_t)E * 4 + 255) & ~(size_t)255;
    float* CG  = (float*)p; p += ((size_t)E * 4 + 255) & ~(size_t)255;
    float* XF  = (float*)p; p += ((size_t)N * FIL * 4 + 255) & ~(size_t)255;
    float* AGG = (float*)p;

    float* h = (float*)d_out;

    init_h_kernel<<<(N * HID + 255) / 256, 256, 0, stream>>>(node_attr, h, N * HID);
    edge_geom_kernel<<<(E + 255) / 256, 256, 0, stream>>>(pos, ei, EL, CG, E);

    const int ntiles = (E + 15) / 16;
    int eblocks = (ntiles + 7) / 8;
    if (eblocks > 4096) eblocks = 4096;            // persistent grid
    const int totalWaves = eblocks * 8;
    const int iters = (ntiles + totalWaves - 1) / totalWaves;   // uniform trip count

    for (int l = 0; l < NLAYERS; ++l) {
        xf_zero_kernel<<<((size_t)N * FIL + 255) / 256, 256, 0, stream>>>(
            h, lin1w + (size_t)l * HID * FIL, XF, AGG, N);
        edge_mlp_kernel<<<eblocks, 256, 0, stream>>>(
            EL, CG, ei,
            dn1w + (size_t)l * NG * FIL, dn1b + (size_t)l * FIL,
            dn2w + (size_t)l * FIL * FIL, dn2b + (size_t)l * FIL,
            XF, AGG, E, ntiles, iters, totalWaves);
        node_update_kernel<<<(N + 7) / 8, 256, 0, stream>>>(
            AGG, lin2w + (size_t)l * FIL * HID, lin2b + (size_t)l * HID,
            linw + (size_t)l * HID * HID, linb + (size_t)l * HID, h, N);
    }
}
